// SKAModule_3590592659972
// MI455X (gfx1250) — matrix operand order repaired and checked by reading, not compiled, not run on hardware
//
#include <hip/hip_runtime.h>
#include <math.h>

// ---- problem constants (match reference) ----
#define D_MODEL  1024
#define N_HEADS  16
#define HEAD_DIM 64
#define RANK     32
#define TSEQ     4096
#define BATCH    4
#define NBH      (BATCH * N_HEADS)   // 64
#define MROWS    (BATCH * TSEQ)      // 16384

typedef __attribute__((ext_vector_type(16))) __bf16 v16bf;
typedef __attribute__((ext_vector_type(8)))  float  v8f;

union Frag16 { v16bf v; int4 q[2]; };

__device__ __forceinline__ unsigned short f2bf(float f) {
    union { float f; unsigned u; } v; v.f = f;
    unsigned r = v.u + 0x7FFFu + ((v.u >> 16) & 1u);   // RNE
    return (unsigned short)(r >> 16);
}

// ---- gfx1250 async global->LDS path (ASYNCcnt), guarded per bridge doc ----
#if __has_builtin(__builtin_amdgcn_global_load_async_to_lds_b128)
#define USE_ASYNC 1
typedef int v4i_raw __attribute__((vector_size(16)));
typedef __attribute__((address_space(1))) v4i_raw* gas_p;
typedef __attribute__((address_space(3))) v4i_raw* las_p;
__device__ __forceinline__ void async_b128(void* lds, const void* g) {
    __builtin_amdgcn_global_load_async_to_lds_b128((gas_p)(g), (las_p)(lds), 0, 0);
}
#if __has_builtin(__builtin_amdgcn_s_wait_asynccnt)
#define WAIT_ASYNC(n) __builtin_amdgcn_s_wait_asynccnt(n)
#else
#define WAIT_ASYNC(n) asm volatile("s_wait_asynccnt " #n ::: "memory")
#endif
#else
#define USE_ASYNC 0
#define WAIT_ASYNC(n)
#endif

// A fragment (16x32 bf16, MxK): lane m=l&15; elems 0-7: K=kh*8+0..7, elems 8-15: K=16+kh*8+0..7 (kh=l>>4)
__device__ __forceinline__ v16bf frag_a(const unsigned short* sm, int lane, int row_off, int k_off, int lds) {
    int m  = row_off + (lane & 15);
    int kb = k_off + ((lane >> 4) << 3);
    Frag16 f;
    f.q[0] = *reinterpret_cast<const int4*>(sm + m * lds + kb);
    f.q[1] = *reinterpret_cast<const int4*>(sm + m * lds + kb + 16);
    return f.v;
}
// B fragment (32x16 bf16, KxN): lane n=l&15; elems 0-7: K=kh*8+0..7, elems 8-15: K=16+kh*8+0..7 (kh=l>>4)
__device__ __forceinline__ v16bf frag_b(const unsigned short* sm, int lane, int n_off, int lds) {
    int n  = n_off + (lane & 15);
    int kb = (lane >> 4) << 3;
    union { v16bf v; unsigned short u[16]; } f;
    #pragma unroll
    for (int s = 0; s < 8; ++s) {
        f.u[s]     = sm[(kb + s) * lds + n];
        f.u[8 + s] = sm[(kb + 16 + s) * lds + n];
    }
    return f.v;
}

#define WMMA_BF16(A, B, C) \
    __builtin_amdgcn_wmma_f32_16x16x32_bf16(false, (A), false, (B), (short)0, (C), false, false)

// ------------------- fp32 -> bf16 weight pre-conversion -------------------
__global__ __launch_bounds__(256) void w2bf_kernel(const float* __restrict__ src,
                                                   unsigned short* __restrict__ dst, int n4) {
    int i = blockIdx.x * 256 + threadIdx.x;
    if (i >= n4) return;
    float4 v = *reinterpret_cast<const float4*>(src + (size_t)i * 4);
    union { unsigned short h[4]; uint2 u; } pk;
    pk.h[0] = f2bf(v.x); pk.h[1] = f2bf(v.y); pk.h[2] = f2bf(v.z); pk.h[3] = f2bf(v.w);
    *reinterpret_cast<uint2*>(dst + (size_t)i * 4) = pk.u;
}

// ------------------- LayerNorm -> bf16 -------------------
__global__ __launch_bounds__(256) void ln_kernel(const float* __restrict__ x,
                                                 const float* __restrict__ sc,
                                                 const float* __restrict__ bi,
                                                 unsigned short* __restrict__ out) {
    __shared__ float red[256];
    int row = blockIdx.x, tid = threadIdx.x;
    const float* xr = x + (size_t)row * D_MODEL;
    float4 v = *reinterpret_cast<const float4*>(xr + tid * 4);
    float s  = v.x + v.y + v.z + v.w;
    float s2 = v.x * v.x + v.y * v.y + v.z * v.z + v.w * v.w;
    red[tid] = s; __syncthreads();
    for (int o = 128; o > 0; o >>= 1) { if (tid < o) red[tid] += red[tid + o]; __syncthreads(); }
    float mean = red[0] * (1.0f / D_MODEL);
    __syncthreads();
    red[tid] = s2; __syncthreads();
    for (int o = 128; o > 0; o >>= 1) { if (tid < o) red[tid] += red[tid + o]; __syncthreads(); }
    float var  = red[0] * (1.0f / D_MODEL) - mean * mean;
    float rstd = rsqrtf(var + 1e-5f);
    float4 g  = *reinterpret_cast<const float4*>(sc + tid * 4);
    float4 b4 = *reinterpret_cast<const float4*>(bi + tid * 4);
    union { unsigned short h[4]; uint2 u; } pk;
    pk.h[0] = f2bf((v.x - mean) * rstd * g.x + b4.x);
    pk.h[1] = f2bf((v.y - mean) * rstd * g.y + b4.y);
    pk.h[2] = f2bf((v.z - mean) * rstd * g.z + b4.z);
    pk.h[3] = f2bf((v.w - mean) * rstd * g.w + b4.w);
    *reinterpret_cast<uint2*>(out + (size_t)row * D_MODEL + tid * 4) = pk.u;
}

// ------------------- Projection GEMM (bf16 x bf16), async double-buffered LDS, scatter to (B,H,T,dim) ----
__global__ __launch_bounds__(256) void gemm_proj_kernel(
        const unsigned short* __restrict__ A,   // [MROWS x 1024] bf16
        const unsigned short* __restrict__ W,   // [1024 x N] bf16
        float* __restrict__ Out,                // (B,H,T,dim) fp32
        int N, int dimShift,
        const int* __restrict__ prefix, int limitP) {
    int P = prefix[0]; P = P < 1 ? 1 : (P > TSEQ - 1 ? TSEQ - 1 : P);
    int rowBase = blockIdx.y * 128;
    if (limitP && (rowBase & (TSEQ - 1)) >= P) return;   // V only needed for prefix
    int colBase = blockIdx.x * 64;
    __shared__ unsigned short sA[2][128 * 40];
    __shared__ unsigned short sB[2][32 * 72];
    int tid = threadIdx.x, lane = tid & 31, wid = tid >> 5;
    int wm = (wid & 3) * 32, wn = (wid >> 2) * 32;
    int ra = tid >> 1, ca = (tid & 1) * 16;
    int rb = tid >> 3, cb = (tid & 7) * 8;
    const unsigned short* gA = A + (size_t)(rowBase + ra) * D_MODEL + ca;
    const unsigned short* gB = W + (size_t)rb * N + colBase + cb;
    const int S = D_MODEL / 32;

    auto stage = [&](int k, int buf) {
#if USE_ASYNC
        async_b128(&sA[buf][ra * 40 + ca],     gA + k * 32);
        async_b128(&sA[buf][ra * 40 + ca + 8], gA + k * 32 + 8);
        async_b128(&sB[buf][rb * 72 + cb],     gB + (size_t)k * 32 * N);
#else
        int4 x0 = *reinterpret_cast<const int4*>(gA + k * 32);
        int4 x1 = *reinterpret_cast<const int4*>(gA + k * 32 + 8);
        int4 y0 = *reinterpret_cast<const int4*>(gB + (size_t)k * 32 * N);
        *reinterpret_cast<int4*>(&sA[buf][ra * 40 + ca])     = x0;
        *reinterpret_cast<int4*>(&sA[buf][ra * 40 + ca + 8]) = x1;
        *reinterpret_cast<int4*>(&sB[buf][rb * 72 + cb])     = y0;
#endif
    };

    v8f z = {0.f,0.f,0.f,0.f,0.f,0.f,0.f,0.f};
    v8f acc[2][2] = {{z, z}, {z, z}};
    stage(0, 0);
    for (int k = 0; k < S; ++k) {
        int cur = k & 1;
        if (k + 1 < S) { stage(k + 1, cur ^ 1); WAIT_ASYNC(3); }
        else           { WAIT_ASYNC(0); }
        __syncthreads();
        v16bf a0 = frag_a(sA[cur], lane, wm,      0, 40);
        v16bf a1 = frag_a(sA[cur], lane, wm + 16, 0, 40);
        v16bf b0 = frag_b(sB[cur], lane, wn,      72);
        v16bf b1 = frag_b(sB[cur], lane, wn + 16, 72);
        acc[0][0] = WMMA_BF16(a0, b0, acc[0][0]);
        acc[0][1] = WMMA_BF16(a0, b1, acc[0][1]);
        acc[1][0] = WMMA_BF16(a1, b0, acc[1][0]);
        acc[1][1] = WMMA_BF16(a1, b1, acc[1][1]);
        __syncthreads();
    }
    int dim = 1 << dimShift, dmask = dim - 1;
    #pragma unroll
    for (int mt = 0; mt < 2; ++mt)
      #pragma unroll
      for (int nt = 0; nt < 2; ++nt)
        #pragma unroll
        for (int i = 0; i < 8; ++i) {
            int row = rowBase + wm + mt * 16 + i + ((lane >> 4) << 3);
            int col = colBase + wn + nt * 16 + (lane & 15);
            int b = row >> 12, t = row & (TSEQ - 1);
            int h = col >> dimShift, j = col & dmask;
            Out[(((size_t)(b * N_HEADS + h)) * TSEQ + t) * dim + j] = acc[mt][nt][i];
        }
}

// ------------------- Prefix stats: G = K'K, M = K[1:]'K[:-1], Cv = V'K (fp32) -------------------
__global__ __launch_bounds__(256) void stats_kernel(
        const float* __restrict__ Kf, const float* __restrict__ Vf,
        float* __restrict__ G, float* __restrict__ Mo, float* __restrict__ Cv,
        const int* __restrict__ prefix) {
    int bh = blockIdx.x, tid = threadIdx.x;
    int P = prefix[0]; P = P < 1 ? 1 : (P > TSEQ - 1 ? TSEQ - 1 : P);
    const float* kp = Kf + (size_t)bh * TSEQ * RANK;
    const float* vp = Vf + (size_t)bh * TSEQ * HEAD_DIM;
    __shared__ float sk[65 * 32];
    __shared__ float sv[64 * 64];
    float gA[4] = {0,0,0,0}, mA[4] = {0,0,0,0}, cA[8] = {0,0,0,0,0,0,0,0};
    int gr[4], gs[4], cd[8], cr[8];
    #pragma unroll
    for (int p = 0; p < 4; ++p) { int idx = tid + p * 256; gr[p] = idx >> 5; gs[p] = idx & 31; }
    #pragma unroll
    for (int q = 0; q < 8; ++q) { int idx = tid + q * 256; cd[q] = idx >> 5; cr[q] = idx & 31; }
    for (int t0 = 0; t0 < P; t0 += 64) {
        int nt = P - t0; if (nt > 64) nt = 64;
        for (int idx = tid; idx < (nt + 1) * 32; idx += 256) {
            int tt = t0 - 1 + (idx >> 5);
            sk[idx] = (tt >= 0) ? kp[(size_t)tt * 32 + (idx & 31)] : 0.0f;
        }
        for (int idx = tid; idx < nt * 64; idx += 256)
            sv[idx] = vp[(size_t)(t0 + (idx >> 6)) * 64 + (idx & 63)];
        __syncthreads();
        for (int l = 0; l < nt; ++l) {
            const float* kc = sk + (l + 1) * 32;
            const float* kq = sk + l * 32;
            #pragma unroll
            for (int p = 0; p < 4; ++p) gA[p] += kc[gr[p]] * kc[gs[p]];
            if (t0 + l >= 1) {
                #pragma unroll
                for (int p = 0; p < 4; ++p) mA[p] += kc[gr[p]] * kq[gs[p]];
            }
            const float* vc = sv + l * 64;
            #pragma unroll
            for (int q = 0; q < 8; ++q) cA[q] += vc[cd[q]] * kc[cr[q]];
        }
        __syncthreads();
    }
    #pragma unroll
    for (int p = 0; p < 4; ++p) { G[(size_t)bh * 1024 + tid + p * 256]  = gA[p];
                                  Mo[(size_t)bh * 1024 + tid + p * 256] = mA[p]; }
    #pragma unroll
    for (int q = 0; q < 8; ++q)   Cv[(size_t)bh * 2048 + tid + q * 256] = cA[q];
}

// ------------------- Per-head solver: chol, A_w, spectral scale, F = Bv L A^4 L^-1 -------------------
__global__ __launch_bounds__(64) void solver_kernel(
        const float* __restrict__ G, const float* __restrict__ Mi,
        const float* __restrict__ Cv, float* __restrict__ F,
        const float* __restrict__ log_ridge, const float* __restrict__ log_gamma) {
    int bh = blockIdx.x, tid = threadIdx.x;
    float ridge = expf(log_ridge[0]);
    float gamma = fminf(expf(log_gamma[0]), 1.0f);
    __shared__ float Ls[32 * 33];
    __shared__ float Ms[32 * 32];
    __shared__ float Aw[32 * 33];
    __shared__ float A2[32 * 33];
    __shared__ float A4[32 * 33];
    __shared__ float CT[32 * 64];
    __shared__ float Eb[64 * 32];
    __shared__ float F1[64 * 32];
    __shared__ float vv[32], uv[32];
    __shared__ float scal;

    for (int idx = tid; idx < 1024; idx += 64) {
        int r = idx >> 5, c = idx & 31;
        Ls[r * 33 + c] = G[(size_t)bh * 1024 + idx] + (r == c ? ridge : 0.0f);
        Ms[idx] = Mi[(size_t)bh * 1024 + idx];
    }
    for (int idx = tid; idx < 2048; idx += 64) {
        int r = idx & 31, d = idx >> 5;
        CT[r * 64 + d] = Cv[(size_t)bh * 2048 + idx];     // Cv is [d][r]
    }
    __syncthreads();
    // Cholesky (in place, lower)
    for (int k = 0; k < 32; ++k) {
        if (tid == 0) Ls[k * 33 + k] = sqrtf(fmaxf(Ls[k * 33 + k], 1e-20f));
        __syncthreads();
        if (tid > k && tid < 32) Ls[tid * 33 + k] /= Ls[k * 33 + k];
        __syncthreads();
        for (int idx = tid; idx < 1024; idx += 64) {
            int i = idx >> 5, j = idx & 31;
            if (i > k && j > k && j <= i) Ls[i * 33 + j] -= Ls[i * 33 + k] * Ls[j * 33 + k];
        }
        __syncthreads();
    }
    // U = L^-1 M (in place, per column)
    if (tid < 32) {
        int c = tid;
        for (int i = 0; i < 32; ++i) {
            float s = Ms[i * 32 + c];
            for (int j = 0; j < i; ++j) s -= Ls[i * 33 + j] * Ms[j * 32 + c];
            Ms[i * 32 + c] = s / Ls[i * 33 + i];
        }
    }
    __syncthreads();
    // A_w = (L^-1 U^T)^T : Aw[c][i] = V[i][c]
    if (tid < 32) {
        int c = tid;
        for (int i = 0; i < 32; ++i) {
            float s = Ms[c * 32 + i];
            for (int j = 0; j < i; ++j) s -= Ls[i * 33 + j] * Aw[c * 33 + j];
            Aw[c * 33 + i] = s / Ls[i * 33 + i];
        }
    }
    __syncthreads();
    // spectral norm via power iteration on A^T A
    if (tid < 32) vv[tid] = 0.1767767f;
    __syncthreads();
    for (int it = 0; it < 32; ++it) {
        if (tid < 32) { float s = 0; for (int j = 0; j < 32; ++j) s += Aw[tid * 33 + j] * vv[j]; uv[tid] = s; }
        __syncthreads();
        if (tid < 32) { float s = 0; for (int j = 0; j < 32; ++j) s += Aw[j * 33 + tid] * uv[j]; vv[tid] = s; }
        __syncthreads();
        if (tid == 0) { float n = 0; for (int j = 0; j < 32; ++j) n += vv[j] * vv[j]; scal = rsqrtf(fmaxf(n, 1e-30f)); }
        __syncthreads();
        if (tid < 32) vv[tid] *= scal;
        __syncthreads();
    }
    if (tid < 32) { float s = 0; for (int j = 0; j < 32; ++j) s += Aw[tid * 33 + j] * vv[j]; uv[tid] = s; }
    __syncthreads();
    if (tid == 0) {
        float n = 0; for (int j = 0; j < 32; ++j) n += uv[j] * uv[j];
        float sigma = fmaxf(sqrtf(n), 1e-8f);
        scal = gamma / fmaxf(sigma, 1.0f);
    }
    __syncthreads();
    for (int idx = tid; idx < 1024; idx += 64) Aw[(idx >> 5) * 33 + (idx & 31)] *= scal;
    __syncthreads();
    // A2 = A*A ; A4 = A2*A2
    for (int idx = tid; idx < 1024; idx += 64) {
        int i = idx >> 5, j = idx & 31; float s = 0;
        for (int k = 0; k < 32; ++k) s += Aw[i * 33 + k] * Aw[k * 33 + j];
        A2[i * 33 + j] = s;
    }
    __syncthreads();
    for (int idx = tid; idx < 1024; idx += 64) {
        int i = idx >> 5, j = idx & 31; float s = 0;
        for (int k = 0; k < 32; ++k) s += A2[i * 33 + k] * A2[k * 33 + j];
        A4[i * 33 + j] = s;
    }
    __syncthreads();
    // Z = L^-T L^-1 Cv^T  (in place in CT, per column d)
    {
        int d = tid;
        for (int i = 0; i < 32; ++i) {
            float s = CT[i * 64 + d];
            for (int j = 0; j < i; ++j) s -= Ls[i * 33 + j] * CT[j * 64 + d];
            CT[i * 64 + d] = s / Ls[i * 33 + i];
        }
        for (int i = 31; i >= 0; --i) {
            float s = CT[i * 64 + d];
            for (int j = i + 1; j < 32; ++j) s -= Ls[j * 33 + i] * CT[j * 64 + d];
            CT[i * 64 + d] = s / Ls[i * 33 + i];
        }
    }
    __syncthreads();
    // E = Bv * L  (Bv[d][k] = Z[k][d])
    for (int idx = tid; idx < 2048; idx += 64) {
        int d = idx >> 5, j = idx & 31; float s = 0;
        for (int k = j; k < 32; ++k) s += CT[k * 64 + d] * Ls[k * 33 + j];
        Eb[d * 32 + j] = s;
    }
    __syncthreads();
    // F1 = E * A4
    for (int idx = tid; idx < 2048; idx += 64) {
        int d = idx >> 5, j = idx & 31; float s = 0;
        for (int k = 0; k < 32; ++k) s += Eb[d * 32 + k] * A4[k * 33 + j];
        F1[d * 32 + j] = s;
    }
    __syncthreads();
    // F * L = F1  -> back substitution (per row d)
    {
        int d = tid;
        for (int j = 31; j >= 0; --j) {
            float s = F1[d * 32 + j];
            for (int k = j + 1; k < 32; ++k) s -= F1[d * 32 + k] * Ls[k * 33 + j];
            F1[d * 32 + j] = s / Ls[j * 33 + j];
        }
    }
    __syncthreads();
    for (int idx = tid; idx < 2048; idx += 64) F[(size_t)bh * 2048 + idx] = F1[idx];
}

// ------------------- WoEff[b, h*32+r, :] = F^T @ Wo[h*64:(h+1)*64, :]  (WMMA) -------------------
__global__ __launch_bounds__(256) void woeff_kernel(
        const float* __restrict__ F, const float* __restrict__ Wo,
        unsigned short* __restrict__ WoEff) {
    int bh = blockIdx.y;
    int b = bh >> 4, h = bh & 15;
    int nbase = blockIdx.x * 128;
    int tid = threadIdx.x, lane = tid & 31, wid = tid >> 5;
    __shared__ unsigned short sFT[32 * 72];   // F^T : [r][d], 32x64
    __shared__ unsigned short sB[64 * 136];   // Wo tile: [k][n], 64x128
    const float* Fb = F + (size_t)bh * 2048;
    for (int idx = tid; idx < 2048; idx += 256) {
        int r = idx & 31, d = idx >> 5;
        sFT[r * 72 + d] = f2bf(Fb[idx]);
    }
    for (int idx = tid; idx < 8192; idx += 256) {
        int r = idx >> 7, c = idx & 127;
        sB[r * 136 + c] = f2bf(Wo[(size_t)(h * 64 + r) * 1024 + nbase + c]);
    }
    __syncthreads();
    int wn = wid * 16;
    v8f z = {0.f,0.f,0.f,0.f,0.f,0.f,0.f,0.f};
    v8f acc0 = z, acc1 = z;
    #pragma unroll
    for (int ks = 0; ks < 64; ks += 32) {
        v16bf a0 = frag_a(sFT, lane, 0,  ks, 72);
        v16bf a1 = frag_a(sFT, lane, 16, ks, 72);
        v16bf bb = frag_b(sB + ks * 136, lane, wn, 136);
        acc0 = WMMA_BF16(a0, bb, acc0);
        acc1 = WMMA_BF16(a1, bb, acc1);
    }
    #pragma unroll
    for (int i = 0; i < 8; ++i) {
        int r0  = i + ((lane >> 4) << 3);
        int col = nbase + wn + (lane & 15);
        WoEff[((size_t)b * 512 + h * 32 + r0) * 1024 + col]      = f2bf(acc0[i]);
        WoEff[((size_t)b * 512 + h * 32 + 16 + r0) * 1024 + col] = f2bf(acc1[i]);
    }
}

// ------------------- pack K (t<P) / Q (t>=P) into bf16 [b*T+t, h*32+r] -------------------
__global__ __launch_bounds__(256) void select_kernel(
        const float* __restrict__ Kf, const float* __restrict__ Qf,
        unsigned short* __restrict__ QK, const int* __restrict__ prefix) {
    int P = prefix[0]; P = P < 1 ? 1 : (P > TSEQ - 1 ? TSEQ - 1 : P);
    size_t g = ((size_t)blockIdx.x * 256 + threadIdx.x) * 4;
    if (g >= (size_t)MROWS * 512) return;
    int col = (int)(g & 511), row = (int)(g >> 9);
    int b = row >> 12, t = row & (TSEQ - 1);
    int h = col >> 5, r = col & 31;
    const float* src = ((t < P) ? Kf : Qf) + ((size_t)(b * N_HEADS + h) * TSEQ + t) * 32 + r;
    union { unsigned short h4[4]; uint2 u; } pk;
    pk.h4[0] = f2bf(src[0]); pk.h4[1] = f2bf(src[1]);
    pk.h4[2] = f2bf(src[2]); pk.h4[3] = f2bf(src[3]);
    *reinterpret_cast<uint2*>(QK + g) = pk.u;
}

// ------------------- final GEMM: out[b] = QKsel[b] @ WoEff[b], x sigmoid(alpha) -------------------
__global__ __launch_bounds__(256) void gemm_out_kernel(
        const unsigned short* __restrict__ QK, const unsigned short* __restrict__ WoEff,
        float* __restrict__ Out, const float* __restrict__ gate) {
    float gscale = 1.0f / (1.0f + expf(-gate[0]));
    int rowBase = blockIdx.y * 128;
    int colBase = blockIdx.x * 64;
    int bidx = rowBase >> 12;
    const unsigned short* Wb = WoEff + (size_t)bidx * 512 * 1024;
    __shared__ unsigned short sA[2][128 * 40];
    __shared__ unsigned short sB[2][32 * 72];
    int tid = threadIdx.x, lane = tid & 31, wid = tid >> 5;
    int wm = (wid & 3) * 32, wn = (wid >> 2) * 32;
    int ra = tid >> 1, ca = (tid & 1) * 16;
    int rb = tid >> 3, cb = (tid & 7) * 8;
    const unsigned short* gA = QK + (size_t)(rowBase + ra) * 512 + ca;
    const unsigned short* gB = Wb + (size_t)rb * 1024 + colBase + cb;
    const int S = 512 / 32;

    auto stage = [&](int k, int buf) {
#if USE_ASYNC
        async_b128(&sA[buf][ra * 40 + ca],     gA + k * 32);
        async_b128(&sA[buf][ra * 40 + ca + 8], gA + k * 32 + 8);
        async_b128(&sB[buf][rb * 72 + cb],     gB + (size_t)k * 32 * 1024);
#else
        int4 x0 = *reinterpret_cast<const int4*>(gA + k * 32);
        int4 x1 = *reinterpret_cast<const int4*>(gA + k * 32 + 8);
        int4 y0 = *reinterpret_cast<const int4*>(gB + (size_t)k * 32 * 1024);
        *reinterpret_cast<int4*>(&sA[buf][ra * 40 + ca])     = x0;
        *reinterpret_cast<int4*>(&sA[buf][ra * 40 + ca + 8]) = x1;
        *reinterpret_cast<int4*>(&sB[buf][rb * 72 + cb])     = y0;
#endif
    };

    v8f z = {0.f,0.f,0.f,0.f,0.f,0.f,0.f,0.f};
    v8f acc[2][2] = {{z, z}, {z, z}};
    stage(0, 0);
    for (int k = 0; k < S; ++k) {
        int cur = k & 1;
        if (k + 1 < S) { stage(k + 1, cur ^ 1); WAIT_ASYNC(3); }
        else           { WAIT_ASYNC(0); }
        __syncthreads();
        v16bf a0 = frag_a(sA[cur], lane, wm,      0, 40);
        v16bf a1 = frag_a(sA[cur], lane, wm + 16, 0, 40);
        v16bf b0 = frag_b(sB[cur], lane, wn,      72);
        v16bf b1 = frag_b(sB[cur], lane, wn + 16, 72);
        acc[0][0] = WMMA_BF16(a0, b0, acc[0][0]);
        acc[0][1] = WMMA_BF16(a0, b1, acc[0][1]);
        acc[1][0] = WMMA_BF16(a1, b0, acc[1][0]);
        acc[1][1] = WMMA_BF16(a1, b1, acc[1][1]);
        __syncthreads();
    }
    #pragma unroll
    for (int mt = 0; mt < 2; ++mt)
      #pragma unroll
      for (int nt = 0; nt < 2; ++nt)
        #pragma unroll
        for (int i = 0; i < 8; ++i) {
            int row = rowBase + wm + mt * 16 + i + ((lane >> 4) << 3);
            int col = colBase + wn + nt * 16 + (lane & 15);
            Out[(size_t)row * 1024 + col] = gscale * acc[mt][nt][i];
        }
}

extern "C" void kernel_launch(void* const* d_in, const int* in_sizes, int n_in,
                              void* d_out, int out_size, void* d_ws, size_t ws_size,
                              hipStream_t stream) {
    (void)in_sizes; (void)n_in; (void)out_size; (void)ws_size;
    const float* hidden     = (const float*)d_in[0];
    const float* Wk         = (const float*)d_in[1];
    const float* Wq         = (const float*)d_in[2];
    const float* Wv         = (const float*)d_in[3];
    const float* Wo         = (const float*)d_in[4];
    const float* ln_scale   = (const float*)d_in[5];
    const float* ln_bias    = (const float*)d_in[6];
    const float* gate_alpha = (const float*)d_in[7];
    const float* log_ridge  = (const float*)d_in[8];
    const float* log_gamma  = (const float*)d_in[9];
    const int*   prefix     = (const int*)d_in[10];

    char* ws = (char*)d_ws;
    size_t off = 0;
    auto take = [&](size_t bytes) -> char* {
        char* p = ws + off; off += (bytes + 255) & ~(size_t)255; return p;
    };
    unsigned short* normed = (unsigned short*)take((size_t)MROWS * D_MODEL * 2);
    float* Kf = (float*)take((size_t)NBH * TSEQ * RANK * 4);
    float* Qf = (float*)take((size_t)NBH * TSEQ * RANK * 4);
    float* Vf = (float*)take((size_t)NBH * TSEQ * HEAD_DIM * 4);
    float* Gb = (float*)take((size_t)NBH * 1024 * 4);
    float* Mb = (float*)take((size_t)NBH * 1024 * 4);
    float* Cb = (float*)take((size_t)NBH * 2048 * 4);
    float* Fb = (float*)take((size_t)NBH * 2048 * 4);
    unsigned short* WoEff = (unsigned short*)take((size_t)BATCH * 512 * 1024 * 2);
    unsigned short* QKs   = (unsigned short*)take((size_t)MROWS * 512 * 2);
    unsigned short* Wkb   = (unsigned short*)take((size_t)D_MODEL * 512 * 2);
    unsigned short* Wqb   = (unsigned short*)take((size_t)D_MODEL * 512 * 2);
    unsigned short* Wvb   = (unsigned short*)take((size_t)D_MODEL * 1024 * 2);

    w2bf_kernel<<<(D_MODEL * 512 / 4 + 255) / 256, 256, 0, stream>>>(Wk, Wkb, D_MODEL * 512 / 4);
    w2bf_kernel<<<(D_MODEL * 512 / 4 + 255) / 256, 256, 0, stream>>>(Wq, Wqb, D_MODEL * 512 / 4);
    w2bf_kernel<<<(D_MODEL * 1024 / 4 + 255) / 256, 256, 0, stream>>>(Wv, Wvb, D_MODEL * 1024 / 4);
    ln_kernel<<<MROWS, 256, 0, stream>>>(hidden, ln_scale, ln_bias, normed);
    gemm_proj_kernel<<<dim3(8, 128),  256, 0, stream>>>(normed, Wkb, Kf, 512, 5, prefix, 0);
    gemm_proj_kernel<<<dim3(8, 128),  256, 0, stream>>>(normed, Wqb, Qf, 512, 5, prefix, 0);
    gemm_proj_kernel<<<dim3(16, 128), 256, 0, stream>>>(normed, Wvb, Vf, 1024, 6, prefix, 1);
    stats_kernel<<<NBH, 256, 0, stream>>>(Kf, Vf, Gb, Mb, Cb, prefix);
    solver_kernel<<<NBH, 64, 0, stream>>>(Gb, Mb, Cb, Fb, log_ridge, log_gamma);
    woeff_kernel<<<dim3(8, NBH), 256, 0, stream>>>(Fb, Wo, WoEff);
    select_kernel<<<(MROWS * 512 / 4) / 256, 256, 0, stream>>>(Kf, Qf, QKs, prefix);
    gemm_out_kernel<<<dim3(16, 128), 256, 0, stream>>>(QKs, WoEff, (float*)d_out, gate_alpha);
}
